// Model_62740882260270
// MI455X (gfx1250) — matrix operand order repaired and checked by reading, not compiled, not run on hardware
//
#include <hip/hip_runtime.h>
#include <hip/hip_bf16.h>
#include <hip/hip_fp16.h>

// ---- problem constants (match reference) ----
#define SEQ  128
#define BAT  64
#define WCH  16
#define WE   300
#define CE   30
#define FN   4
#define CK   3
#define CC   512
#define NT   20
#define NL   3
#define KPAD 448          // 420 padded to multiple of 32
#define NWORDS (SEQ*BAT)  // 8192

typedef _Float16 half_t;
typedef __attribute__((ext_vector_type(16))) _Float16 v16h;
typedef __attribute__((ext_vector_type(8)))  _Float16 v8h;
typedef __attribute__((ext_vector_type(8)))  float    v8f;

union frag { v16h v; v8h h[2]; };

static __device__ __forceinline__ v8f wmma_f16(const frag& a, const frag& b, v8f c) {
    return __builtin_amdgcn_wmma_f32_16x16x32_f16(false, a.v, false, b.v,
                                                  (short)0, c, false, false);
}

// =====================================================================
// K0: weight prep (fp32 -> f16, transposed layouts) + zero pad rows
// =====================================================================
__global__ void prep_kernel(const float* __restrict__ conv_w,
                            const float* __restrict__ lin_w,
                            half_t* __restrict__ w16,        // [l][k][o][i]
                            half_t* __restrict__ lin_wT16,   // [c][f] (448)
                            half_t* __restrict__ h16_0,
                            half_t* __restrict__ h16_1) {
    const int NW16 = NL * 3 * 1024 * CC;       // 4,718,592
    const int NLIN = CC * KPAD;                // 229,376
    const int NPAD = 2 * BAT * 2 * CC;         // 131,072
    const int total = NW16 + NLIN + NPAD;
    for (int idx = blockIdx.x * blockDim.x + threadIdx.x; idx < total;
         idx += gridDim.x * blockDim.x) {
        if (idx < NW16) {
            int i = idx & 511;
            int tmp = idx >> 9;
            int o = tmp & 1023;
            int tmp2 = tmp >> 10;
            int k = tmp2 % 3;
            int l = tmp2 / 3;
            w16[idx] = (half_t)conv_w[(((size_t)(l * 1024 + o) * CC + i) * 3) + k];
        } else if (idx < NW16 + NLIN) {
            int j = idx - NW16;
            int f = j % KPAD, c = j / KPAD;
            lin_wT16[(size_t)c * KPAD + f] =
                (f < WE + CE * FN) ? (half_t)lin_w[(size_t)f * CC + c] : (half_t)0.f;
        } else {
            int j = idx - NW16 - NLIN;
            int c = j & 511;
            int r = (j >> 9) & 1;        // 0 -> row 0, 1 -> row S+1
            int b = (j >> 10) & 63;
            int p = j >> 16;
            half_t* h16 = p ? h16_1 : h16_0;
            h16[((size_t)b * (SEQ + 2) + (r ? (SEQ + 1) : 0)) * CC + c] = (half_t)0.f;
        }
    }
}

// =====================================================================
// K1: word-emb gather + char-CNN + maxpool -> feat16 (8192 x 448) f16
// =====================================================================
__global__ void __launch_bounds__(128)
feat_kernel(const int* __restrict__ words, const int* __restrict__ chars,
            const float* __restrict__ word_emb, const float* __restrict__ char_emb,
            const float* __restrict__ ccw, const float* __restrict__ ccb,
            half_t* __restrict__ feat16) {
    int n = blockIdx.x;          // feat row index (s*B + b)
    int s = n / BAT, b = n % BAT;
    int t = threadIdx.x;
    __shared__ int   cid[WCH];
    __shared__ float chs[WCH][CE];
    if (t < WCH) cid[t] = chars[((size_t)(b * SEQ + s)) * WCH + t];
    __syncthreads();
    for (int j = t; j < WCH * CE; j += blockDim.x) {
        int w = j / CE, g = j % CE;
        chs[w][g] = char_emb[(size_t)cid[w] * CE + g];
    }
    __syncthreads();
    int wid = words[n];
    for (int j = t; j < WE; j += blockDim.x)
        feat16[(size_t)n * KPAD + j] = (half_t)word_emb[(size_t)wid * WE + j];
    if (t < CE * FN) {
        int o = t, g = o >> 2;
        float k0 = ccw[o * 3 + 0], k1 = ccw[o * 3 + 1], k2 = ccw[o * 3 + 2];
        float bia = ccb[o];
        float mx = -1e30f;
        #pragma unroll
        for (int p = 0; p < WCH - CK + 1; ++p) {
            float v = chs[p][g] * k0 + chs[p + 1][g] * k1 + chs[p + 2][g] * k2 + bia;
            mx = fmaxf(mx, v);
        }
        feat16[(size_t)n * KPAD + WE + o] = (half_t)mx;
    }
    if (t >= 100 && t < 128)     // zero K pad cols 420..447
        feat16[(size_t)n * KPAD + 420 + (t - 100)] = (half_t)0.f;
}

// =====================================================================
// K2: linear GEMM via WMMA, 2M x 2N register tile per wave
//   A = lin_wT16 (M=c 512, K=448), B = feat16^T (K, N=8192)
// =====================================================================
__global__ void __launch_bounds__(256)
lin_gemm_kernel(const half_t* __restrict__ feat16, const half_t* __restrict__ lin_wT16,
                const float* __restrict__ lin_b,
                float* __restrict__ h32, half_t* __restrict__ h16) {
    int gw   = blockIdx.x * (blockDim.x >> 5) + (threadIdx.x >> 5);
    int lane = threadIdx.x & 31;
    const int MT2 = CC / 32;                // 16 double-M tiles
    int mt = gw % MT2;
    int nt = gw / MT2;
    if (nt >= NWORDS / 32) return;
    int hi  = lane >> 4;
    int l15 = lane & 15;
    const half_t* A0 = lin_wT16 + (size_t)(mt * 32 + l15) * KPAD;
    const half_t* A1 = A0 + (size_t)16 * KPAD;
    const half_t* B0 = feat16 + (size_t)(nt * 32 + l15) * KPAD;
    const half_t* B1 = B0 + (size_t)16 * KPAD;
    v8f acc[2][2] = {};
    #pragma unroll
    for (int kb = 0; kb < KPAD; kb += 32) {
        frag a0, a1, b0, b1;
        a0.h[0] = *(const v8h*)(A0 + kb + hi * 8);
        a0.h[1] = *(const v8h*)(A0 + kb + 16 + hi * 8);
        a1.h[0] = *(const v8h*)(A1 + kb + hi * 8);
        a1.h[1] = *(const v8h*)(A1 + kb + 16 + hi * 8);
        b0.h[0] = *(const v8h*)(B0 + kb + hi * 8);
        b0.h[1] = *(const v8h*)(B0 + kb + 16 + hi * 8);
        b1.h[0] = *(const v8h*)(B1 + kb + hi * 8);
        b1.h[1] = *(const v8h*)(B1 + kb + 16 + hi * 8);
        acc[0][0] = wmma_f16(a0, b0, acc[0][0]);
        acc[1][0] = wmma_f16(a1, b0, acc[1][0]);
        acc[0][1] = wmma_f16(a0, b1, acc[0][1]);
        acc[1][1] = wmma_f16(a1, b1, acc[1][1]);
    }
    #pragma unroll
    for (int ni = 0; ni < 2; ++ni) {
        int ncol = nt * 32 + ni * 16 + l15;
        int s = ncol / BAT, b = ncol % BAT;
        #pragma unroll
        for (int mi = 0; mi < 2; ++mi) {
            #pragma unroll
            for (int r = 0; r < 8; ++r) {
                int c = mt * 32 + mi * 16 + r + hi * 8;
                float v = acc[mi][ni][r] + lin_b[c];
                h32[((size_t)b * SEQ + s) * CC + c] = v;
                h16[((size_t)b * (SEQ + 2) + s + 1) * CC + c] = (half_t)v;
            }
        }
    }
}

// =====================================================================
// K3: gated conv layer as WMMA GEMM, fused GLU + residual
//   per wave: paired M tiles (o, o+512) x 2 N tiles (32 time steps)
//   A frags reused across 2 columns, B frags across both gates
// =====================================================================
__global__ void __launch_bounds__(256)
conv_gemm_kernel(const half_t* __restrict__ w16l, const float* __restrict__ conv_bl,
                 const half_t* __restrict__ h16_in, half_t* __restrict__ h16_out,
                 float* __restrict__ h32) {
    int gw   = blockIdx.x * (blockDim.x >> 5) + (threadIdx.x >> 5);
    int lane = threadIdx.x & 31;
    const int MT = CC / 16;                 // 32 paired M tiles
    int mt = gw % MT;
    int nt = gw / MT;                       // covers n in [nt*32, nt*32+32)
    if (nt >= NWORDS / 32) return;
    int hi  = lane >> 4;
    int l15 = lane & 15;
    int obase = mt * 16;
    int arow  = obase + l15;
    int n0 = nt * 32 + l15;
    int b  = n0 >> 7;                       // 32 | 128 -> tile within one batch row
    int t0 = n0 & 127;
    v8f acc0[2] = {}, acc1[2] = {};         // [ni] for gates a (o) and g (o+512)
    #pragma unroll
    for (int k = 0; k < 3; ++k) {
        const half_t* A0 = w16l + ((size_t)k * 1024 + arow) * CC;
        const half_t* A1 = A0 + (size_t)512 * CC;
        const half_t* Bc0 = h16_in + ((size_t)b * (SEQ + 2) + t0 + k) * CC;
        const half_t* Bc1 = Bc0 + (size_t)16 * CC;
        #pragma unroll
        for (int kb = 0; kb < CC; kb += 32) {
            frag a0, a1, b0, b1;
            b0.h[0] = *(const v8h*)(Bc0 + kb + hi * 8);
            b0.h[1] = *(const v8h*)(Bc0 + kb + 16 + hi * 8);
            b1.h[0] = *(const v8h*)(Bc1 + kb + hi * 8);
            b1.h[1] = *(const v8h*)(Bc1 + kb + 16 + hi * 8);
            a0.h[0] = *(const v8h*)(A0 + kb + hi * 8);
            a0.h[1] = *(const v8h*)(A0 + kb + 16 + hi * 8);
            a1.h[0] = *(const v8h*)(A1 + kb + hi * 8);
            a1.h[1] = *(const v8h*)(A1 + kb + 16 + hi * 8);
            acc0[0] = wmma_f16(a0, b0, acc0[0]);
            acc1[0] = wmma_f16(a1, b0, acc1[0]);
            acc0[1] = wmma_f16(a0, b1, acc0[1]);
            acc1[1] = wmma_f16(a1, b1, acc1[1]);
        }
    }
    const float scale = 0.70710678118654752f;   // sqrt(0.5)
    #pragma unroll
    for (int ni = 0; ni < 2; ++ni) {
        int t = t0 + ni * 16;
        #pragma unroll
        for (int r = 0; r < 8; ++r) {
            int c = obase + r + hi * 8;
            float a = acc0[ni][r] + conv_bl[c];
            float g = acc1[ni][r] + conv_bl[c + 512];
            float sig = 1.0f / (1.0f + expf(-g));
            size_t hidx = ((size_t)b * SEQ + t) * CC + c;
            float hn = (a * sig + h32[hidx]) * scale;
            h32[hidx] = hn;
            h16_out[((size_t)b * (SEQ + 2) + t + 1) * CC + c] = (half_t)hn;
        }
    }
}

// =====================================================================
// K4: fc -> emissions (S,B,NT) f32
// =====================================================================
__global__ void fc_kernel(const float* __restrict__ h32, const float* __restrict__ fc_w,
                          const float* __restrict__ fc_b, float* __restrict__ emis) {
    int idx = blockIdx.x * blockDim.x + threadIdx.x;
    if (idx >= NWORDS * NT) return;
    int j = idx % NT;
    int n = idx / NT;            // s*B + b
    int s = n / BAT, b = n % BAT;
    const float* hrow = h32 + ((size_t)b * SEQ + s) * CC;
    float acc = fc_b[j];
    for (int c = 0; c < CC; ++c) acc += hrow[c] * fc_w[c * NT + j];
    emis[(size_t)n * NT + j] = acc;
}

// =====================================================================
// K5: CRF Viterbi + NLL, one wave per batch item
// =====================================================================
__global__ void __launch_bounds__(32)
crf_kernel(const float* __restrict__ emis, const int* __restrict__ words,
           const int* __restrict__ tags,
           const float* __restrict__ crf_start, const float* __restrict__ crf_end,
           const float* __restrict__ crf_trans,
           unsigned char* __restrict__ bp, float* __restrict__ out_tags,
           float* __restrict__ loss_part) {
    int b = blockIdx.x;
    int t = threadIdx.x;
    __shared__ float trans[NT * NT];
    __shared__ float alpha[NT], nalpha[NT], score[NT], nscore[NT];
    __shared__ float cs[NT], cef[NT];
    for (int j = t; j < NT * NT; j += 32) trans[j] = crf_trans[j];
    if (t < NT) { cs[t] = crf_start[t]; cef[t] = crf_end[t]; }
    __syncthreads();
    if (t < NT) {
        float e0 = emis[(size_t)b * NT + t];
        alpha[t] = cs[t] + e0;
        score[t] = cs[t] + e0;
    }
    __syncthreads();
    for (int s = 1; s < SEQ; ++s) {
        bool m = (words[s * BAT + b] != 0);
        if (t < NT) {
            float e = emis[(size_t)(s * BAT + b) * NT + t];
            float mx = -1e30f, bestv = -1e30f;
            int besti = 0;
            for (int i = 0; i < NT; ++i) {
                float va = alpha[i] + trans[i * NT + t];
                mx = fmaxf(mx, va);
                float vs = score[i] + trans[i * NT + t];
                if (vs > bestv) { bestv = vs; besti = i; }
            }
            float ssum = 0.f;
            for (int i = 0; i < NT; ++i)
                ssum += expf(alpha[i] + trans[i * NT + t] - mx);
            nalpha[t] = m ? (mx + logf(ssum) + e) : alpha[t];
            nscore[t] = m ? (bestv + e) : score[t];
            bp[((size_t)(s - 1) * BAT + b) * NT + t] = (unsigned char)(m ? besti : t);
        }
        __syncthreads();
        if (t < NT) { alpha[t] = nalpha[t]; score[t] = nscore[t]; }
        __syncthreads();
    }
    if (t == 0) {
        float mx = -1e30f;
        for (int j = 0; j < NT; ++j) mx = fmaxf(mx, alpha[j] + cef[j]);
        float ssum = 0.f;
        for (int j = 0; j < NT; ++j) ssum += expf(alpha[j] + cef[j] - mx);
        float logz = mx + logf(ssum);
        float bv = -1e30f; int btag = 0;
        for (int j = 0; j < NT; ++j) {
            float v = score[j] + cef[j];
            if (v > bv) { bv = v; btag = j; }
        }
        int tag = btag;
        for (int s = SEQ - 1; s >= 1; --s) {
            out_tags[(size_t)s * BAT + b] = (float)tag;
            tag = bp[((size_t)(s - 1) * BAT + b) * NT + tag];
        }
        out_tags[b] = (float)tag;
        // NLL numerator
        float num = cs[tags[b]] + emis[(size_t)b * NT + tags[b]];
        int nmask = (words[b] != 0) ? 1 : 0;
        for (int s = 1; s < SEQ; ++s) {
            int tp = tags[(s - 1) * BAT + b], tc = tags[s * BAT + b];
            float m = (words[s * BAT + b] != 0) ? 1.f : 0.f;
            num += (trans[tp * NT + tc] + emis[(size_t)(s * BAT + b) * NT + tc]) * m;
            nmask += (words[s * BAT + b] != 0) ? 1 : 0;
        }
        int last_idx = nmask - 1;
        num += cef[tags[last_idx * BAT + b]];
        loss_part[b] = -(num - logz);
    }
}

// K6: deterministic fixed-order loss reduction
__global__ void loss_reduce_kernel(const float* __restrict__ loss_part, float* __restrict__ out) {
    if (threadIdx.x == 0 && blockIdx.x == 0) {
        float s = 0.f;
        for (int b = 0; b < BAT; ++b) s += loss_part[b];
        out[NWORDS] = s;
    }
}

// =====================================================================
extern "C" void kernel_launch(void* const* d_in, const int* in_sizes, int n_in,
                              void* d_out, int out_size, void* d_ws, size_t ws_size,
                              hipStream_t stream) {
    const int*   words     = (const int*)d_in[0];
    const int*   chars     = (const int*)d_in[1];
    const int*   tags      = (const int*)d_in[2];
    const float* word_emb  = (const float*)d_in[3];
    const float* char_emb  = (const float*)d_in[4];
    const float* ccw       = (const float*)d_in[5];
    const float* ccb       = (const float*)d_in[6];
    const float* lin_w     = (const float*)d_in[7];
    const float* lin_b     = (const float*)d_in[8];
    const float* conv_w    = (const float*)d_in[9];
    const float* conv_b    = (const float*)d_in[10];
    const float* fc_w      = (const float*)d_in[11];
    const float* fc_b      = (const float*)d_in[12];
    const float* crf_start = (const float*)d_in[13];
    const float* crf_end   = (const float*)d_in[14];
    const float* crf_trans = (const float*)d_in[15];
    float* out = (float*)d_out;

    char* ws = (char*)d_ws;
    size_t off = 0;
    auto alloc = [&](size_t bytes) -> void* {
        void* p = ws + off;
        off = (off + bytes + 255) & ~(size_t)255;
        return p;
    };
    half_t* w16      = (half_t*)alloc((size_t)NL * 3 * 1024 * CC * 2);
    half_t* lin_wT16 = (half_t*)alloc((size_t)CC * KPAD * 2);
    half_t* feat16   = (half_t*)alloc((size_t)NWORDS * KPAD * 2);
    half_t* h16a     = (half_t*)alloc((size_t)BAT * (SEQ + 2) * CC * 2);
    half_t* h16b     = (half_t*)alloc((size_t)BAT * (SEQ + 2) * CC * 2);
    float*  h32      = (float*)alloc((size_t)BAT * SEQ * CC * 4);
    float*  emis     = (float*)alloc((size_t)NWORDS * NT * 4);
    unsigned char* bp = (unsigned char*)alloc((size_t)(SEQ - 1) * BAT * NT);
    float*  loss_part = (float*)alloc(BAT * 4);

    // K0: weight prep + pad-row zeroing
    {
        int total = NL * 3 * 1024 * CC + CC * KPAD + 2 * BAT * 2 * CC;
        int blocks = (total + 255) / 256;
        prep_kernel<<<blocks, 256, 0, stream>>>(conv_w, lin_w, w16, lin_wT16, h16a, h16b);
    }
    // K1: features
    feat_kernel<<<NWORDS, 128, 0, stream>>>(words, chars, word_emb, char_emb, ccw, ccb, feat16);
    // K2: linear GEMM (WMMA)  -> h32, h16a
    {
        int waves = (CC / 32) * (NWORDS / 32);   // 4096
        lin_gemm_kernel<<<waves / 8, 256, 0, stream>>>(feat16, lin_wT16, lin_b, h32, h16a);
    }
    // K3: three gated conv layers (WMMA), h16 ping-pong
    {
        int waves = (CC / 16) * (NWORDS / 32);   // 8192
        half_t* pin = h16a; half_t* pout = h16b;
        for (int l = 0; l < NL; ++l) {
            conv_gemm_kernel<<<waves / 8, 256, 0, stream>>>(
                w16 + (size_t)l * 3 * 1024 * CC, conv_b + (size_t)l * 1024,
                pin, pout, h32);
            half_t* tmp = pin; pin = pout; pout = tmp;
        }
    }
    // K4: emissions
    fc_kernel<<<(NWORDS * NT + 255) / 256, 256, 0, stream>>>(h32, fc_w, fc_b, emis);
    // K5: CRF viterbi + NLL
    crf_kernel<<<BAT, 32, 0, stream>>>(emis, words, tags, crf_start, crf_end, crf_trans,
                                       bp, out, loss_part);
    // K6: deterministic loss sum
    loss_reduce_kernel<<<1, 32, 0, stream>>>(loss_part, out);
}
